// MultiHeadAttentionClassical_65481071397077
// MI455X (gfx1250) — matrix operand order repaired and checked by reading, not compiled, not run on hardware
//
#include <hip/hip_runtime.h>
#include <hip/hip_bf16.h>

// ---------------------------------------------------------------------------
// Multi-head attention, bf16 WMMA + async-LDS pipeline for gfx1250 (MI455X).
//   B=2, S=2048, E=1024, H=16, Dk=64.
//   ws layout (u16 elements): xb | WqT WkT WvT WoT (N x K) | Qb Kb Vtg | ctxb
//   Vtg is V stored transposed as (B, H, Dk, S) so attention tiles are
//   straight row copies (no LDS scatter anywhere in hot loops).
// ---------------------------------------------------------------------------

typedef unsigned short u16;
typedef unsigned int   u32;

typedef __attribute__((ext_vector_type(16))) __bf16 v16bf;
typedef __attribute__((ext_vector_type(8)))  __bf16 v8bf;
typedef __attribute__((ext_vector_type(8)))  float  v8f;

#define BATCH 2
#define SEQ   2048
#define EMB   1024
#define HEADS 16
#define DK    64

// ---------------- helpers -------------------------------------------------

__device__ __forceinline__ u16 f2bf(float f) {
    u32 u = __builtin_bit_cast(u32, f);
    u += 0x7FFFu + ((u >> 16) & 1u);      // round-to-nearest-even
    return (u16)(u >> 16);
}

__device__ __forceinline__ v8f wmma_bf16(v16bf a, v16bf b, v8f c) {
    return __builtin_amdgcn_wmma_f32_16x16x32_bf16(false, a, false, b,
                                                   (short)0, c, false, false);
}

// Async global->LDS 16B copy (per lane). VDST holds the wave-relative LDS
// byte address: low 32 bits of the generic pointer ARE the LDS offset
// (flat-aperture rule).  Tracked by ASYNCcnt.
__device__ __forceinline__ void async_copy_b128(void* lds_ptr, const void* gptr) {
    u32 lds_addr = (u32)(__UINTPTR_TYPE__)lds_ptr;
    asm volatile("global_load_async_to_lds_b128 %0, %1, off"
                 :: "v"(lds_addr), "v"(gptr)
                 : "memory");
}
__device__ __forceinline__ void wait_async_le4() {
    asm volatile("s_wait_asynccnt 4" ::: "memory");
}
__device__ __forceinline__ void wait_async_0() {
    asm volatile("s_wait_asynccnt 0" ::: "memory");
}

// A-matrix fragment (16x32 bf16).  lane%16 -> row, lane/16 selects K halves:
//   elems 0..7  <- K = k0 + 8*(lane>>4) + i ; elems 8..15 <- +16
__device__ __forceinline__ v16bf load_a_frag(const u16* lds, int pitch,
                                             int m0, int k0) {
    int lane = threadIdx.x & 31;
    int m  = m0 + (lane & 15);
    int kh = (lane >> 4) << 3;
    const v8bf h0 = *reinterpret_cast<const v8bf*>(lds + m * pitch + k0 + kh);
    const v8bf h1 = *reinterpret_cast<const v8bf*>(lds + m * pitch + k0 + 16 + kh);
    return __builtin_shufflevector(h0, h1, 0,1,2,3,4,5,6,7,8,9,10,11,12,13,14,15);
}

// B-matrix fragment (32x16 bf16) from an (N x K) LDS tile:
//   lane%16 -> column n, elems 0..7 <- K = k0 + 8*(lane>>4) + i ; elems 8..15 <- +16
__device__ __forceinline__ v16bf load_b_frag(const u16* lds, int pitch,
                                             int n0, int k0) {
    int lane = threadIdx.x & 31;
    int n  = n0 + (lane & 15);
    int kg = (lane >> 4) << 3;
    const v8bf h0 = *reinterpret_cast<const v8bf*>(lds + n * pitch + k0 + kg);
    const v8bf h1 = *reinterpret_cast<const v8bf*>(lds + n * pitch + k0 + 16 + kg);
    return __builtin_shufflevector(h0, h1, 0,1,2,3,4,5,6,7,8,9,10,11,12,13,14,15);
}

// ---------------- conversions ---------------------------------------------

__global__ void mha_f32_to_bf16(const float* __restrict__ in,
                                u16* __restrict__ out, int n) {
    int i = blockIdx.x * blockDim.x + threadIdx.x;
    int stride = gridDim.x * blockDim.x;
    for (; i < n; i += stride) out[i] = f2bf(in[i]);
}

// W is (K x N) fp32 row-major; emit bf16 transposed (N x K) so GEMM B-tiles
// become contiguous copies.
__global__ void mha_w_to_bf16_t(const float* __restrict__ in,
                                u16* __restrict__ out, int K, int N) {
    int i = blockIdx.x * blockDim.x + threadIdx.x;
    int stride = gridDim.x * blockDim.x;
    int total = K * N;
    for (; i < total; i += stride) {
        int n = i / K;
        int k = i - n * K;
        out[i] = f2bf(in[(size_t)k * N + n]);
    }
}

// ---------------- bf16 GEMM: C = scale * (A @ B^T_stored) -----------------
// A: MxK row-major bf16.  Bt: NxK row-major bf16 (pre-transposed weights).
// 128x128 block, 256 threads (8 waves), 2x4 WMMA tiles per wave,
// double-buffered LDS staged with async global->LDS copies.
// OMODE: 0 = fp32 row-major, 1 = bf16 row-major, 2 = bf16 into (B,H,Dk,S).

template <int OMODE>
__global__ __launch_bounds__(256)
void mha_gemm_bf16(const u16* __restrict__ A, const u16* __restrict__ BT,
                   void* __restrict__ Cp, int M, int N, int K, float scale) {
    constexpr int PA   = 40;               // 32 + 8 pad; 80B rows (16B aligned)
    constexpr int TILE = 128 * PA;
    __shared__ __align__(16) u16 At[2 * TILE];   // double-buffered 128x32
    __shared__ __align__(16) u16 Bt[2 * TILE];

    const int tid    = threadIdx.x;
    const int lane   = tid & 31;
    const int wave   = tid >> 5;
    const int wave_m = wave >> 1;          // 0..3 : 32-row strip
    const int wave_n = wave & 1;           // 0..1 : 64-col strip
    const int m0 = blockIdx.y * 128;
    const int n0 = blockIdx.x * 128;

    v8f acc[2][4];
    #pragma unroll
    for (int i = 0; i < 2; ++i)
        #pragma unroll
        for (int j = 0; j < 4; ++j) acc[i][j] = (v8f){};

    // Stage issue: 512 chunks of 8 bf16 per tile -> 2 A + 2 B async ops/thread
    auto issue_stage = [&](int k0, int buf) {
        u16* Ab = At + buf * TILE;
        u16* Bb = Bt + buf * TILE;
        #pragma unroll
        for (int c = 0; c < 2; ++c) {
            int linear = c * 256 + tid;
            int r  = linear >> 2;          // 0..127
            int ch = linear & 3;           // 0..3  (x8 K elements)
            async_copy_b128(&Ab[r * PA + ch * 8],
                            A  + (size_t)(m0 + r) * K + k0 + ch * 8);
            async_copy_b128(&Bb[r * PA + ch * 8],
                            BT + (size_t)(n0 + r) * K + k0 + ch * 8);
        }
    };

    const int T = K / 32;
    issue_stage(0, 0);
    for (int t = 0; t < T; ++t) {
        if (t + 1 < T) {
            issue_stage((t + 1) * 32, (t + 1) & 1);
            wait_async_le4();              // stage t landed; t+1 in flight
        } else {
            wait_async_0();
        }
        __syncthreads();

        const u16* Ab = At + (t & 1) * TILE;
        const u16* Bb = Bt + (t & 1) * TILE;
        v16bf afrag[2];
        #pragma unroll
        for (int i = 0; i < 2; ++i)
            afrag[i] = load_a_frag(Ab, PA, wave_m * 32 + i * 16, 0);
        #pragma unroll
        for (int j = 0; j < 4; ++j) {
            v16bf bfrag = load_b_frag(Bb, PA, wave_n * 64 + j * 16, 0);
            #pragma unroll
            for (int i = 0; i < 2; ++i)
                acc[i][j] = wmma_bf16(afrag[i], bfrag, acc[i][j]);
        }
        __syncthreads();
    }

    // Epilogue: VGPR r holds row (r + 8*(lane>>4)), col = lane&15
    const int rsub = (lane >> 4) << 3;
    #pragma unroll
    for (int i = 0; i < 2; ++i) {
        #pragma unroll
        for (int j = 0; j < 4; ++j) {
            #pragma unroll
            for (int r = 0; r < 8; ++r) {
                int row = m0 + wave_m * 32 + i * 16 + rsub + r;
                int col = n0 + wave_n * 64 + j * 16 + (lane & 15);
                float v = acc[i][j][r] * scale;
                if (OMODE == 0) {
                    reinterpret_cast<float*>(Cp)[(size_t)row * N + col] = v;
                } else if (OMODE == 1) {
                    reinterpret_cast<u16*>(Cp)[(size_t)row * N + col] = f2bf(v);
                } else {                   // (B,H,Dk,S) transposed store for V
                    int bb = row >> 11;    // row / SEQ
                    int ss = row & (SEQ - 1);
                    int hh = col >> 6;     // col / DK
                    int dd = col & (DK - 1);
                    reinterpret_cast<u16*>(Cp)[
                        (((size_t)bb * HEADS + hh) * DK + dd) * SEQ + ss] = f2bf(v);
                }
            }
        }
    }
}

// ---------------- flash attention -----------------------------------------
// One workgroup per (b, h, 64-query tile). Keys/values streamed in 64-blocks
// via async global->LDS copies; scores never leave LDS. Q carries 1/sqrt(Dk).
// V comes pre-transposed as (B,H,Dk,S), so every tile is a straight copy.

__global__ __launch_bounds__(256)
void mha_flash_attn(const u16* __restrict__ Qb, const u16* __restrict__ Kb,
                    const u16* __restrict__ Vtg, u16* __restrict__ ctx) {
    constexpr int P  = 72;                 // bf16 pitch (144B, 16B aligned)
    constexpr int PF = 68;                 // f32 pitch  (272B, 16B aligned)
    __shared__ __align__(16) u16  Qt[64 * P];    // 64 q rows x 64 dims
    __shared__ __align__(16) u16  Kt[64 * P];    // 64 keys  x 64 dims (Bt for Q@K^T)
    __shared__ __align__(16) u16  Vt[64 * P];    // 64 dims  x 64 keys (Bt for P@V)
    __shared__ __align__(16) u16  Pt[64 * P];    // probs bf16 (A for P@V)
    __shared__ __align__(16) float Sc[64 * PF];  // scores f32
    __shared__ float m_s[64], l_s[64], a_s[64];

    const int b  = blockIdx.z;
    const int h  = blockIdx.y;
    const int q0 = blockIdx.x * 64;
    const int tid    = threadIdx.x;
    const int lane   = tid & 31;
    const int wave   = tid >> 5;
    const int wave_m = wave >> 1;          // 16-row group of queries
    const int wave_n = wave & 1;           // 32-col strip

    const u16* Qg = Qb  + ((size_t)(b * SEQ + q0)) * EMB + h * DK;
    const u16* Vg = Vtg + ((size_t)(b * HEADS + h)) * DK * SEQ;

    // Q tile once: 64 rows x 8 chunks = 512 -> 2 async ops per thread
    #pragma unroll
    for (int c = 0; c < 2; ++c) {
        int linear = c * 256 + tid;
        int r  = linear >> 3;
        int ch = linear & 7;
        async_copy_b128(&Qt[r * P + ch * 8], Qg + (size_t)r * EMB + ch * 8);
    }
    if (tid < 64) { m_s[tid] = -1.0e30f; l_s[tid] = 0.0f; }

    v8f oacc[2];
    oacc[0] = (v8f){};
    oacc[1] = (v8f){};
    wait_async_0();
    __syncthreads();

    for (int kb = 0; kb < SEQ / 64; ++kb) {
        const u16* Kg = Kb + ((size_t)(b * SEQ + kb * 64)) * EMB + h * DK;
        // K tile: rows = keys (dims contiguous).  V tile: rows = dims
        // (keys contiguous in the transposed global layout).
        #pragma unroll
        for (int c = 0; c < 2; ++c) {
            int linear = c * 256 + tid;
            int r  = linear >> 3;          // 0..63
            int ch = linear & 7;
            async_copy_b128(&Kt[r * P + ch * 8], Kg + (size_t)r * EMB + ch * 8);
            async_copy_b128(&Vt[r * P + ch * 8],
                            Vg + (size_t)r * SEQ + kb * 64 + ch * 8);
        }
        wait_async_0();
        __syncthreads();

        // S = Q @ K^T
        v8f sacc[2];
        sacc[0] = (v8f){};
        sacc[1] = (v8f){};
        #pragma unroll
        for (int ks = 0; ks < 2; ++ks) {
            v16bf aq = load_a_frag(Qt, P, wave_m * 16, ks * 32);
            #pragma unroll
            for (int j = 0; j < 2; ++j) {
                v16bf bk = load_b_frag(Kt, P, (wave_n * 2 + j) * 16, ks * 32);
                sacc[j] = wmma_bf16(aq, bk, sacc[j]);
            }
        }
        {
            const int rsub = (lane >> 4) << 3;
            #pragma unroll
            for (int j = 0; j < 2; ++j)
                #pragma unroll
                for (int r = 0; r < 8; ++r) {
                    int row = wave_m * 16 + rsub + r;
                    int col = (wave_n * 2 + j) * 16 + (lane & 15);
                    Sc[row * PF + col] = sacc[j][r];
                }
        }
        __syncthreads();

        // online softmax, one thread per query row
        if (tid < 64) {
            const int row = tid;
            float mold = m_s[row];
            float mx = mold;
            for (int c = 0; c < 64; ++c) mx = fmaxf(mx, Sc[row * PF + c]);
            float alpha = __expf(mold - mx);
            float sum = 0.0f;
            for (int c = 0; c < 64; ++c) {
                float p = __expf(Sc[row * PF + c] - mx);
                sum += p;
                Pt[row * P + c] = f2bf(p);
            }
            l_s[row] = l_s[row] * alpha + sum;
            m_s[row] = mx;
            a_s[row] = alpha;
        }
        __syncthreads();

        // rescale running O, then O += P @ V
        {
            const int rowb = wave_m * 16 + ((lane >> 4) << 3);
            float al[8];
            #pragma unroll
            for (int r = 0; r < 8; ++r) al[r] = a_s[rowb + r];
            #pragma unroll
            for (int j = 0; j < 2; ++j)
                #pragma unroll
                for (int r = 0; r < 8; ++r) oacc[j][r] *= al[r];
        }
        #pragma unroll
        for (int ks = 0; ks < 2; ++ks) {
            v16bf ap = load_a_frag(Pt, P, wave_m * 16, ks * 32);
            #pragma unroll
            for (int j = 0; j < 2; ++j) {
                v16bf bv = load_b_frag(Vt, P, (wave_n * 2 + j) * 16, ks * 32);
                oacc[j] = wmma_bf16(ap, bv, oacc[j]);
            }
        }
        __syncthreads();   // protect Kt/Vt/Sc/Pt before next block
    }

    // normalize by row sum, store bf16 context
    const int rowb = wave_m * 16 + ((lane >> 4) << 3);
    #pragma unroll
    for (int j = 0; j < 2; ++j) {
        #pragma unroll
        for (int r = 0; r < 8; ++r) {
            int row  = rowb + r;
            float li = 1.0f / l_s[row];
            int d    = (wave_n * 2 + j) * 16 + (lane & 15);
            ctx[((size_t)(b * SEQ + q0 + row)) * EMB + h * DK + d] =
                f2bf(oacc[j][r] * li);
        }
    }
}

// ---------------- host launch ---------------------------------------------

extern "C" void kernel_launch(void* const* d_in, const int* in_sizes, int n_in,
                              void* d_out, int out_size, void* d_ws, size_t ws_size,
                              hipStream_t stream) {
    (void)in_sizes; (void)n_in; (void)out_size; (void)ws_size;

    const float* x  = (const float*)d_in[0];
    const float* Wq = (const float*)d_in[1];
    const float* Wk = (const float*)d_in[2];
    const float* Wv = (const float*)d_in[3];
    const float* Wo = (const float*)d_in[4];

    const size_t SE = (size_t)BATCH * SEQ * EMB;   // 4,194,304
    const size_t EE = (size_t)EMB * EMB;           // 1,048,576
    u16* ws  = (u16*)d_ws;
    u16* xb  = ws;
    u16* wqt = xb  + SE;
    u16* wkt = wqt + EE;
    u16* wvt = wkt + EE;
    u16* wot = wvt + EE;
    u16* qb  = wot + EE;
    u16* kbf = qb  + SE;
    u16* vtg = kbf + SE;     // V transposed: (B, H, Dk, S)
    u16* ctx = vtg + SE;

    // 1) fp32 -> bf16 (x) and fp32 -> bf16 transposed (weights, one time)
    mha_f32_to_bf16<<<(int)((SE + 255) / 256), 256, 0, stream>>>(x, xb, (int)SE);
    mha_w_to_bf16_t<<<(int)((EE + 255) / 256), 256, 0, stream>>>(Wq, wqt, EMB, EMB);
    mha_w_to_bf16_t<<<(int)((EE + 255) / 256), 256, 0, stream>>>(Wk, wkt, EMB, EMB);
    mha_w_to_bf16_t<<<(int)((EE + 255) / 256), 256, 0, stream>>>(Wv, wvt, EMB, EMB);
    mha_w_to_bf16_t<<<(int)((EE + 255) / 256), 256, 0, stream>>>(Wo, wot, EMB, EMB);

    const int M = BATCH * SEQ;   // 4096
    const int N = EMB;           // 1024
    const int K = EMB;           // 1024
    dim3 ggrid(N / 128, M / 128);

    // 2) projections (1/sqrt(Dk) folded into Q; V stored transposed)
    mha_gemm_bf16<1><<<ggrid, 256, 0, stream>>>(xb, wqt, qb,  M, N, K, 0.125f);
    mha_gemm_bf16<1><<<ggrid, 256, 0, stream>>>(xb, wkt, kbf, M, N, K, 1.0f);
    mha_gemm_bf16<2><<<ggrid, 256, 0, stream>>>(xb, wvt, vtg, M, N, K, 1.0f);

    // 3) flash attention -> bf16 context
    dim3 agrid(SEQ / 64, HEADS, BATCH);
    mha_flash_attn<<<agrid, 256, 0, stream>>>(qb, kbf, vtg, ctx);

    // 4) output projection -> fp32
    mha_gemm_bf16<0><<<ggrid, 256, 0, stream>>>(ctx, wot, d_out, M, N, K, 1.0f);
}
